// GinoGridToMesh_49744311222712
// MI455X (gfx1250) — matrix operand order repaired and checked by reading, not compiled, not run on hardware
//
#include <hip/hip_runtime.h>
#include <hip/hip_bf16.h>

typedef __attribute__((ext_vector_type(16))) _Float16 v16h;
typedef __attribute__((ext_vector_type(8)))  _Float16 v8h;
typedef __attribute__((ext_vector_type(8)))  float    v8f;

#define NWAVES 8
#define BLOCK  256

__device__ __forceinline__ float gelu_exact(float x) {
  return 0.5f * x * (1.0f + erff(x * 0.70710678118654752f));
}

// A fragment (16x32 f16, M x K): lane r=L&15 is row M=r; sel=L>>4 selects the
// K interleave: halves = K[sel*8 .. sel*8+7] then K[sel*8+16 .. sel*8+23].
__device__ __forceinline__ v16h load_fragA(const _Float16* base, int stride) {
  int lane = threadIdx.x & 31;
  int r = lane & 15, sel = lane >> 4;
  const _Float16* p = base + r * stride + sel * 8;
  v8h lo = *(const v8h*)(p);
  v8h hi = *(const v8h*)(p + 16);
  v16h f;
#pragma unroll
  for (int i = 0; i < 8; ++i) { f[i] = lo[i]; f[i + 8] = hi[i]; }
  return f;
}

// B fragment (32x16 f16, K x N) from W^T stored [N,K] row-major:
// lane holds column n=L&15; halves = K[sel*8 .. sel*8+7] then K[sel*8+16 .. sel*8+23].
__device__ __forceinline__ v16h load_fragB(const _Float16* base, int stride) {
  int lane = threadIdx.x & 31;
  int n = lane & 15, sel = lane >> 4;
  const _Float16* p = base + n * stride + sel * 8;
  v8h lo = *(const v8h*)(p);
  v8h hi = *(const v8h*)(p + 16);
  v16h f;
#pragma unroll
  for (int i = 0; i < 8; ++i) { f[i] = lo[i]; f[i + 8] = hi[i]; }
  return f;
}

// 64-row GEMM over LDS input:  out[64,N] = act(in[64,K] @ W + b)
// Each active wave owns NI contiguous 16-wide N tiles; k is the only loop so
// A fragments (4 per k-step) are loaded from LDS and consumed immediately —
// nothing is loop-invariant, nothing spills.
// MODE 0: store f16 to lds_out   MODE 1: store f16 to global rows (guarded)
// MODE 2: atomicAdd f32 scatter into segment sums via qrow[]
template<int NI, int MODE, bool GELU>
__device__ __forceinline__ void mlp_gemm(
    const _Float16* lds_in, int stride_in,
    const _Float16* WT, int K, const float* bias, int Ntiles,
    _Float16* lds_out, int stride_out,
    _Float16* gout, int gN, long long row_base, long long row_max,
    float* seg, const int* qrow, int segN)
{
  int wave = threadIdx.x >> 5;
  int lane = threadIdx.x & 31;
  int r = lane & 15, sel = lane >> 4;
  int Ktiles = K >> 5;
  int active = Ntiles / NI;           // waves with work (wave-uniform test)
  if (wave >= active) return;
  int ni0 = wave * NI;
  const _Float16* wb = WT + (size_t)ni0 * 16 * K;

  v8f acc[NI][4] = {};
  for (int k = 0; k < Ktiles; ++k) {
    v16h a[4];
#pragma unroll
    for (int mi = 0; mi < 4; ++mi)
      a[mi] = load_fragA(lds_in + mi * 16 * stride_in + k * 32, stride_in);
#pragma unroll
    for (int j = 0; j < NI; ++j) {
      v16h bf = load_fragB(wb + (size_t)j * 16 * K + k * 32, K);
#pragma unroll
      for (int mi = 0; mi < 4; ++mi)
        acc[j][mi] = __builtin_amdgcn_wmma_f32_16x16x32_f16(
            false, a[mi], false, bf, (short)0, acc[j][mi], false, false);
    }
  }

#pragma unroll
  for (int j = 0; j < NI; ++j) {
    int ni = ni0 + j;
    float bv = bias[ni * 16 + r];
    int col = ni * 16 + r;
#pragma unroll
    for (int mi = 0; mi < 4; ++mi) {
#pragma unroll
      for (int rr = 0; rr < 8; ++rr) {
        float v = acc[j][mi][rr] + bv;
        if (GELU) v = gelu_exact(v);
        int row = mi * 16 + rr + sel * 8;
        if (MODE == 0) {
          lds_out[row * stride_out + col] = (_Float16)v;
        } else if (MODE == 1) {
          long long gr = row_base + row;
          if (gr < row_max) gout[gr * (long long)gN + col] = (_Float16)v;
        } else {
          long long er = row_base + row;
          if (er < row_max) atomicAdd(seg + (size_t)qrow[row] * segN + col, v);
        }
      }
    }
  }
}

// ---- weight convert+transpose: src f32 [K,N] row-major -> dst f16 [N,K] ----
__global__ void wconv_kernel(const float* __restrict__ src, _Float16* __restrict__ dst,
                             int K, int N) {
  long long i = (long long)blockIdx.x * BLOCK + threadIdx.x;
  long long total = (long long)K * N;
  if (i < total) {
    long long n = i / K, k = i % K;
    dst[i] = (_Float16)src[k * (long long)N + n];
  }
}

// ---- zero fill ----
__global__ void zero_f32(float* p, long long n) {
  long long i = (long long)blockIdx.x * BLOCK + threadIdx.x;
  if (i < n) p[i] = 0.0f;
}

// ---- sincos positional embedding -> f16 [NQ,192] ----
__global__ void qe_kernel(const float* __restrict__ qp, _Float16* __restrict__ qeh,
                          long long total) {
  long long i = (long long)blockIdx.x * BLOCK + threadIdx.x;
  if (i >= total) return;
  long long q = i / 192;
  int j = (int)(i % 192);
  int d = j >> 6;          // which of 3 dims
  int jj = j & 63;         // position within 64-wide block
  int w = jj & 31;
  // omega = 10000^(-2w/64) = 2^(-(2w/64)*log2(10000))
  float omega = exp2f(-((float)(2 * w) / 64.0f) * 13.287712379549449f);
  float a = qp[q * 3 + d] * omega;
  float v = (jj < 32) ? __sinf(a) : __cosf(a);
  qeh[i] = (_Float16)v;
}

// ---- proj: xf_h[M,192] = f16(x[M,192] @ proj_w + b) ----
__global__ void proj_kernel(const float* __restrict__ x, const float* __restrict__ b,
                            const _Float16* __restrict__ WT, _Float16* __restrict__ out,
                            long long M) {
  extern __shared__ char smem[];
  _Float16* sIn = (_Float16*)smem;  // 64 x 200 (padded)
  int tid = threadIdx.x;
  long long base = (long long)blockIdx.x * 64;
  for (int idx = tid; idx < 64 * 24; idx += BLOCK) {
    int row = idx / 24, c = idx % 24;
    long long gr = base + row;
    float4 f0 = {0.f, 0.f, 0.f, 0.f}, f1 = {0.f, 0.f, 0.f, 0.f};
    if (gr < M) {
      const float4* p = (const float4*)(x + gr * 192 + c * 8);
      f0 = p[0]; f1 = p[1];
    }
    _Float16* d = sIn + row * 200 + c * 8;
    d[0] = (_Float16)f0.x; d[1] = (_Float16)f0.y; d[2] = (_Float16)f0.z; d[3] = (_Float16)f0.w;
    d[4] = (_Float16)f1.x; d[5] = (_Float16)f1.y; d[6] = (_Float16)f1.z; d[7] = (_Float16)f1.w;
  }
  __syncthreads();
  mlp_gemm<2, 1, false>(sIn, 200, WT, 192, b, 12,
                        nullptr, 0, out, 192, base, M, nullptr, nullptr, 0);
}

// ---- fused 3-layer edge MLP + atomic segment scatter ----
__global__ void edge_kernel(const long long* __restrict__ edges,
                            const _Float16* __restrict__ xfh, const _Float16* __restrict__ qeh,
                            const _Float16* __restrict__ W0T, const float* __restrict__ b0,
                            const _Float16* __restrict__ W1T, const float* __restrict__ b1,
                            const _Float16* __restrict__ W2T, const float* __restrict__ b2,
                            float* __restrict__ seg, float* __restrict__ cnt, long long NE) {
  extern __shared__ char smem[];
  _Float16* bufA = (_Float16*)smem;        // 64 x 392
  _Float16* bufB = bufA + 64 * 392;        // 64 x 392
  int* qarr = (int*)(bufB + 64 * 392);     // 64
  int* garr = qarr + 64;                   // 64
  int tid = threadIdx.x;
  long long base = (long long)blockIdx.x * 64;

  if (tid < 64) {
    long long e = base + tid;
    int q = 0, g = 0;
    if (e < NE) {
      q = (int)edges[2 * e];
      g = (int)edges[2 * e + 1];
      atomicAdd(cnt + q, 1.0f);
    }
    qarr[tid] = q; garr[tid] = g;
  }
  __syncthreads();

  // gather h0 = concat(xf[g], qe[q]) -> bufA, 48 chunks of 8 halfs per row
  for (int idx = tid; idx < 64 * 48; idx += BLOCK) {
    int row = idx / 48, c = idx % 48;
    const _Float16* src = (c < 24)
        ? (xfh + (size_t)garr[row] * 192 + c * 8)
        : (qeh + (size_t)qarr[row] * 192 + (c - 24) * 8);
    *(v8h*)(bufA + row * 392 + c * 8) = *(const v8h*)src;
  }
  __syncthreads();

  // layer 0: 384 -> 384, gelu   (24 N-tiles, 3 per wave)
  mlp_gemm<3, 0, true>(bufA, 392, W0T, 384, b0, 24,
                       bufB, 392, nullptr, 0, 0, 0, nullptr, nullptr, 0);
  __syncthreads();
  // layer 1: 384 -> 384, gelu
  mlp_gemm<3, 0, true>(bufB, 392, W1T, 384, b1, 24,
                       bufA, 392, nullptr, 0, 0, 0, nullptr, nullptr, 0);
  __syncthreads();
  // layer 2: 384 -> 192, linear, atomic scatter (12 N-tiles, 2 per wave, 6 waves)
  mlp_gemm<2, 2, false>(bufA, 392, W2T, 384, b2, 12,
                        nullptr, 0, nullptr, 0, base, NE, seg, qarr, 192);
}

// ---- segment mean -> f16 ----
__global__ void mean_kernel(const float* __restrict__ seg, const float* __restrict__ cnt,
                            _Float16* __restrict__ mh, long long total) {
  long long i = (long long)blockIdx.x * BLOCK + threadIdx.x;
  if (i < total) {
    long long q = i / 192;
    float c = fmaxf(cnt[q], 1.0f);
    mh[i] = (_Float16)(seg[i] / c);
  }
}

// ---- prediction head: gelu(m @ P0 + b0) @ P1 + b1 -> out[NQ,4] ----
__global__ void pred_kernel(const _Float16* __restrict__ mh,
                            const _Float16* __restrict__ P0T, const float* __restrict__ pb0,
                            const float* __restrict__ P1, const float* __restrict__ pb1,
                            float* __restrict__ out, long long NQ) {
  extern __shared__ char smem[];
  _Float16* sIn = (_Float16*)smem;   // 64 x 200
  _Float16* sT  = sIn + 64 * 200;    // 64 x 200
  int tid = threadIdx.x;
  long long base = (long long)blockIdx.x * 64;
  for (int idx = tid; idx < 64 * 24; idx += BLOCK) {
    int row = idx / 24, c = idx % 24;
    long long gr = base + row;
    v8h v = {};
    if (gr < NQ) v = *(const v8h*)(mh + gr * 192 + c * 8);
    *(v8h*)(sIn + row * 200 + c * 8) = v;
  }
  __syncthreads();
  mlp_gemm<2, 0, true>(sIn, 200, P0T, 192, pb0, 12,
                       sT, 200, nullptr, 0, 0, 0, nullptr, nullptr, 0);
  __syncthreads();
  int row = tid >> 2, o = tid & 3;
  long long gr = base + row;
  if (gr < NQ) {
    float acc = pb1[o];
    for (int k = 0; k < 192; ++k)
      acc += (float)sT[row * 200 + k] * P1[k * 4 + o];
    out[gr * 4 + o] = acc;
  }
}

static inline long long cdiv(long long a, long long b) { return (a + b - 1) / b; }

extern "C" void kernel_launch(void* const* d_in, const int* in_sizes, int n_in,
                              void* d_out, int out_size, void* d_ws, size_t ws_size,
                              hipStream_t stream) {
  const float*     x      = (const float*)d_in[0];
  const float*     qpos   = (const float*)d_in[1];
  const long long* edges  = (const long long*)d_in[2];
  const float*     proj_w = (const float*)d_in[3];
  const float*     proj_b = (const float*)d_in[4];
  const float*     m0w    = (const float*)d_in[5];
  const float*     m0b    = (const float*)d_in[6];
  const float*     m1w    = (const float*)d_in[7];
  const float*     m1b    = (const float*)d_in[8];
  const float*     m2w    = (const float*)d_in[9];
  const float*     m2b    = (const float*)d_in[10];
  const float*     p0w    = (const float*)d_in[11];
  const float*     p0b    = (const float*)d_in[12];
  const float*     p1w    = (const float*)d_in[13];
  const float*     p1b    = (const float*)d_in[14];
  float* out = (float*)d_out;

  long long M  = (long long)in_sizes[0] / 192;   // B*S grid points
  long long NQ = (long long)in_sizes[1] / 3;     // queries
  long long NE = (long long)in_sizes[2] / 2;     // edges

  // ---- workspace partition (256B aligned) ----
  char* w = (char*)d_ws;
  auto take = [&](size_t bytes) { void* p = (void*)w; w += (bytes + 255) & ~(size_t)255; return p; };
  _Float16* projT = (_Float16*)take((size_t)192 * 192 * 2);
  _Float16* W0T   = (_Float16*)take((size_t)384 * 384 * 2);
  _Float16* W1T   = (_Float16*)take((size_t)384 * 384 * 2);
  _Float16* W2T   = (_Float16*)take((size_t)192 * 384 * 2);
  _Float16* P0T   = (_Float16*)take((size_t)192 * 192 * 2);
  _Float16* xfh   = (_Float16*)take((size_t)M  * 192 * 2);
  _Float16* qeh   = (_Float16*)take((size_t)NQ * 192 * 2);
  _Float16* mh    = (_Float16*)take((size_t)NQ * 192 * 2);
  float*    seg   = (float*)take((size_t)(NQ * 192 + NQ) * 4);  // sums + counts contiguous
  float*    cnt   = seg + NQ * 192;

  // weight transpose+convert (src [K,N] -> dst f16 [N,K])
  wconv_kernel<<<dim3((unsigned)cdiv(192LL * 192, BLOCK)), BLOCK, 0, stream>>>(proj_w, projT, 192, 192);
  wconv_kernel<<<dim3((unsigned)cdiv(384LL * 384, BLOCK)), BLOCK, 0, stream>>>(m0w, W0T, 384, 384);
  wconv_kernel<<<dim3((unsigned)cdiv(384LL * 384, BLOCK)), BLOCK, 0, stream>>>(m1w, W1T, 384, 384);
  wconv_kernel<<<dim3((unsigned)cdiv(384LL * 192, BLOCK)), BLOCK, 0, stream>>>(m2w, W2T, 384, 192);
  wconv_kernel<<<dim3((unsigned)cdiv(192LL * 192, BLOCK)), BLOCK, 0, stream>>>(p0w, P0T, 192, 192);

  // zero segment sums + counts
  zero_f32<<<dim3((unsigned)cdiv(NQ * 193, BLOCK)), BLOCK, 0, stream>>>(seg, NQ * 193);

  // query sincos embedding
  qe_kernel<<<dim3((unsigned)cdiv(NQ * 192, BLOCK)), BLOCK, 0, stream>>>(qpos, qeh, NQ * 192);

  // grid feature projection (WMMA)
  proj_kernel<<<dim3((unsigned)cdiv(M, 64)), BLOCK, (size_t)64 * 200 * 2, stream>>>(
      x, proj_b, projT, xfh, M);

  // fused edge MLP + scatter (WMMA x3)
  size_t edge_smem = (size_t)64 * 392 * 2 * 2 + 64 * 2 * 4;
  edge_kernel<<<dim3((unsigned)cdiv(NE, 64)), BLOCK, edge_smem, stream>>>(
      edges, xfh, qeh, W0T, m0b, W1T, m1b, W2T, m2b, seg, cnt, NE);

  // segment mean
  mean_kernel<<<dim3((unsigned)cdiv(NQ * 192, BLOCK)), BLOCK, 0, stream>>>(seg, cnt, mh, NQ * 192);

  // prediction head (WMMA + tiny fp32 dot)
  pred_kernel<<<dim3((unsigned)cdiv(NQ, 64)), BLOCK, (size_t)64 * 200 * 2 * 2, stream>>>(
      mh, P0T, p0b, p1w, p1b, out, NQ);
}
